// HawkBlock_10505490006123
// MI455X (gfx1250) — matrix operand order repaired and checked by reading, not compiled, not run on hardware
//
#include <hip/hip_runtime.h>
#include <hip/hip_bf16.h>
#include <math.h>

// ---------------------------------------------------------------------------
// Problem constants (from the reference)
// ---------------------------------------------------------------------------
#define BATCH   4
#define SEQ     2048
#define DIMC    1024
#define HIDC    2048          // DIM * EXP
#define GHIDC   2048          // DIM * GMLP_EXP
#define KCONV   4
#define MROWS   (BATCH * SEQ) // 8192
#define ALPHA_LOG_SCALE (-8.0f)

// ---------------------------------------------------------------------------
// WMMA vector types (wave32, CDNA5)
// ---------------------------------------------------------------------------
typedef __attribute__((ext_vector_type(16))) __bf16 v16bf;
typedef __attribute__((ext_vector_type(8)))  __bf16 v8bf;
typedef __attribute__((ext_vector_type(8)))  float  v8f;

// ---------------------------------------------------------------------------
// Small device helpers
// ---------------------------------------------------------------------------
__device__ __forceinline__ float gelu_exact(float x) {
    return 0.5f * x * (1.0f + erff(x * 0.70710678118654752440f));
}
__device__ __forceinline__ float sigmoidf_(float x) {
    return 1.0f / (1.0f + __expf(-x));
}
// LDS byte address of a __shared__ object = low 32 bits of its flat address
// (flat LDS aperture is SHARED_BASE[63:32] | offset[31:0], ISA 10.2)
__device__ __forceinline__ unsigned ldsOff(const void* p) {
    return (unsigned)(unsigned long long)p;
}

// ---------------------------------------------------------------------------
// fp32 -> bf16 conversion (weights & activations)
// ---------------------------------------------------------------------------
__global__ __launch_bounds__(256) void f2bf_kernel(const float* __restrict__ in,
                                                   __bf16* __restrict__ out,
                                                   long long n) {
    long long i = (long long)blockIdx.x * 256 + threadIdx.x;
    if (i < n) out[i] = (__bf16)in[i];
}

// ---------------------------------------------------------------------------
// RMSNorm (reference flavor: gamma * sqrt(D) * x / ||x||) -> bf16
// one 256-thread block per row
// ---------------------------------------------------------------------------
__global__ __launch_bounds__(256) void rmsnorm_kernel(const float* __restrict__ x,
                                                      const float* __restrict__ gamma,
                                                      __bf16* __restrict__ out,
                                                      int D) {
    __shared__ float red[256];
    long long row = blockIdx.x;
    const float* xr = x + row * (long long)D;
    float s = 0.0f;
    for (int i = threadIdx.x; i < D; i += 256) { float v = xr[i]; s += v * v; }
    red[threadIdx.x] = s;
    __syncthreads();
    for (int st = 128; st > 0; st >>= 1) {
        if (threadIdx.x < st) red[threadIdx.x] += red[threadIdx.x + st];
        __syncthreads();
    }
    float scale = rsqrtf(red[0]) * sqrtf((float)D);
    for (int i = threadIdx.x; i < D; i += 256)
        out[row * (long long)D + i] = (__bf16)(xr[i] * scale * gamma[i]);
}

// ---------------------------------------------------------------------------
// Tiled bf16 WMMA GEMM:  C[M,N] = A[M,K] * W[N,K]^T  (+bias[N]) (+res[M,N])
// block tile 128x128, BK=32, 8 waves (each wave: 32x64 = 2x4 WMMA frags).
// Global->LDS via CDNA5 async-to-LDS DMA (ASYNCcnt), double-buffered so the
// k+1 tile streams in while the 8 WMMAs of tile k execute. One barrier/iter.
// ---------------------------------------------------------------------------
#define BM 128
#define BN 128
#define BK 32
#define LDR 40   // LDS row stride in bf16 (80B = 20 banks -> conflict-free)

__global__ __launch_bounds__(256)
void gemm_bf16_wmma_kernel(const __bf16* __restrict__ A,
                           const __bf16* __restrict__ W,
                           const float* __restrict__ bias,   // nullable
                           const float* __restrict__ res,    // nullable
                           float* __restrict__ C,
                           int M, int N, int K) {
    __shared__ __bf16 As[2 * BM * LDR];
    __shared__ __bf16 Bs[2 * BN * LDR];

    const int tid     = threadIdx.x;
    const int lane    = tid & 31;
    const int wid     = tid >> 5;      // 0..7
    const int wm      = wid >> 1;      // 0..3  (M direction, 32 rows each)
    const int wn      = wid & 1;       // 0..1  (N direction, 64 cols each)
    const int laneCol = lane & 15;
    const int laneHi  = lane >> 4;     // 0 or 1

    const int blockM = blockIdx.y * BM;
    const int blockN = blockIdx.x * BN;

    // loader assignment: thread -> rows (tid>>2, tid>>2+64), 16B chunk (tid&3)
    const int ldRow = tid >> 2;        // 0..63
    const int ldCk  = tid & 3;         // 0..3
    const unsigned sA0 = ldsOff(&As[ldRow * LDR + ldCk * 8]);
    const unsigned sB0 = ldsOff(&Bs[ldRow * LDR + ldCk * 8]);

    auto issueLoads = [&](int buf, int k0) {
        const __bf16* a0 = A + (long long)(blockM + ldRow)      * K + k0 + ldCk * 8;
        const __bf16* a1 = A + (long long)(blockM + ldRow + 64) * K + k0 + ldCk * 8;
        const __bf16* b0 = W + (long long)(blockN + ldRow)      * K + k0 + ldCk * 8;
        const __bf16* b1 = W + (long long)(blockN + ldRow + 64) * K + k0 + ldCk * 8;
        unsigned dA0 = sA0 + buf * (BM * LDR * 2);
        unsigned dA1 = dA0 + 64 * LDR * 2;
        unsigned dB0 = sB0 + buf * (BN * LDR * 2);
        unsigned dB1 = dB0 + 64 * LDR * 2;
        asm volatile("global_load_async_to_lds_b128 %0, %1, off"
                     :: "v"(dA0), "v"((unsigned long long)a0) : "memory");
        asm volatile("global_load_async_to_lds_b128 %0, %1, off"
                     :: "v"(dA1), "v"((unsigned long long)a1) : "memory");
        asm volatile("global_load_async_to_lds_b128 %0, %1, off"
                     :: "v"(dB0), "v"((unsigned long long)b0) : "memory");
        asm volatile("global_load_async_to_lds_b128 %0, %1, off"
                     :: "v"(dB1), "v"((unsigned long long)b1) : "memory");
    };

    v8f acc[2][4] = {};

    const int nIter = K / BK;
    issueLoads(0, 0);

    for (int it = 0; it < nIter; ++it) {
        const int cur = it & 1;
        // my async writes into buffer `cur` done:
        asm volatile("s_wait_asynccnt 0x0" ::: "memory");
        // everyone's writes done; everyone finished reading buffer 1-cur
        // (their fragment ds_loads completed before their WMMAs last iter):
        __syncthreads();
        // stream tile it+1 into the other buffer; overlaps the WMMAs below
        if (it + 1 < nIter) issueLoads(1 - cur, (it + 1) * BK);

        const __bf16* as = &As[cur * BM * LDR];
        const __bf16* bs = &Bs[cur * BN * LDR];

        // ---- A fragments (ISA layout: lanes<16 K{0..7,16..23}; lanes>=16 K{8..15,24..31})
        v16bf af[2];
        #pragma unroll
        for (int i = 0; i < 2; ++i) {
            int r = wm * 32 + i * 16 + laneCol;
            v8bf lo = *(const v8bf*)(&as[r * LDR + laneHi * 8]);
            v8bf hi = *(const v8bf*)(&as[r * LDR + 16 + laneHi * 8]);
            af[i] = __builtin_shufflevector(lo, hi, 0,1,2,3,4,5,6,7,8,9,10,11,12,13,14,15);
        }
        // ---- B fragments (lanes<16: K{0..7,16..23} of col; lanes>=16: K{8..15,24..31})
        v16bf bfr[4];
        #pragma unroll
        for (int j = 0; j < 4; ++j) {
            int r = wn * 64 + j * 16 + laneCol;
            v8bf lo = *(const v8bf*)(&bs[r * LDR + laneHi * 8]);
            v8bf hi = *(const v8bf*)(&bs[r * LDR + 16 + laneHi * 8]);
            bfr[j] = __builtin_shufflevector(lo, hi, 0,1,2,3,4,5,6,7,8,9,10,11,12,13,14,15);
        }

        // ---- 8 WMMAs per wave per K-step ----
        #pragma unroll
        for (int i = 0; i < 2; ++i)
            #pragma unroll
            for (int j = 0; j < 4; ++j)
                acc[i][j] = __builtin_amdgcn_wmma_f32_16x16x32_bf16(
                    false, af[i], false, bfr[j], (short)0, acc[i][j], false, false);
    }

    // ---- epilogue: C/D layout VGPR r -> row r (lanes 0-15) / row r+8 (lanes 16-31)
    #pragma unroll
    for (int i = 0; i < 2; ++i) {
        #pragma unroll
        for (int j = 0; j < 4; ++j) {
            #pragma unroll
            for (int r = 0; r < 8; ++r) {
                int gr = blockM + wm * 32 + i * 16 + laneHi * 8 + r;
                int gc = blockN + wn * 64 + j * 16 + laneCol;
                float v = acc[i][j][r];
                if (bias) v += bias[gc];
                if (res)  v += res[(long long)gr * N + gc];
                C[(long long)gr * N + gc] = v;
            }
        }
    }
}

// ---------------------------------------------------------------------------
// Depthwise causal conv (K=4) on xc = z[..., HID:2*HID]; writes f32 + bf16
// ---------------------------------------------------------------------------
__global__ __launch_bounds__(256)
void conv_kernel(const float* __restrict__ z,
                 const float* __restrict__ conv_w,
                 const float* __restrict__ conv_b,
                 float* __restrict__ xc,
                 __bf16* __restrict__ xcb) {
    long long idx = (long long)blockIdx.x * 256 + threadIdx.x;
    if (idx >= (long long)MROWS * HIDC) return;
    int c = (int)(idx % HIDC);
    long long bt = idx / HIDC;
    int t = (int)(bt % SEQ);
    long long brow = bt - t;          // b*T
    float acc = conv_b[c];
    #pragma unroll
    for (int k = 0; k < KCONV; ++k) {
        int tt = t - (KCONV - 1) + k;
        if (tt >= 0)
            acc += conv_w[c * KCONV + k] *
                   z[(brow + tt) * (long long)(2 * HIDC) + HIDC + c];
    }
    xc[idx]  = acc;
    xcb[idx] = (__bf16)acc;
}

// ---------------------------------------------------------------------------
// Fused RG-LRU: gates -> alpha/beta -> sequential scan over T ->
// a = bf16(gelu(gate) * h).   One thread per (b, channel).
// ---------------------------------------------------------------------------
__global__ __launch_bounds__(256)
void scan_kernel(const float* __restrict__ g,     // [M, 2*HID]
                 const float* __restrict__ xc,    // [M, HID]
                 const float* __restrict__ z,     // gate in z[..., :HID]
                 const float* __restrict__ forget_base,
                 __bf16* __restrict__ a_out) {    // [M, HID]
    int idx = blockIdx.x * 256 + threadIdx.x;
    if (idx >= BATCH * HIDC) return;
    int c = idx % HIDC;
    int b = idx / HIDC;
    float sp   = log1pf(__expf(forget_base[c]));   // softplus
    float coef = ALPHA_LOG_SCALE * sp;             // -8 * softplus(base)
    float h = 0.0f;
    long long base = (long long)b * SEQ;
    #pragma unroll 4
    for (int t = 0; t < SEQ; ++t) {
        long long o4 = (base + t) * (long long)(2 * HIDC);
        long long o2 = (base + t) * (long long)HIDC;
        float forget = g[o4 + c];
        float inp    = g[o4 + HIDC + c];
        float xcv    = xc[o2 + c];
        float gate   = z[o4 + c];
        float alpha  = __expf(coef * sigmoidf_(forget));
        float beta   = sqrtf(1.0f - alpha * alpha + 1e-6f);
        float xs     = beta * sigmoidf_(inp) * xcv;
        h = alpha * h + xs;
        a_out[o2 + c] = (__bf16)(gelu_exact(gate) * h);
    }
}

// ---------------------------------------------------------------------------
// Gated-GeLU for the gMLP: mact = bf16(gelu(gz[:, :GHID]) * gz[:, GHID:])
// ---------------------------------------------------------------------------
__global__ __launch_bounds__(256)
void gmlp_gate_kernel(const float* __restrict__ gz, __bf16* __restrict__ mact) {
    long long idx = (long long)blockIdx.x * 256 + threadIdx.x;
    if (idx >= (long long)MROWS * GHIDC) return;
    int j = (int)(idx % GHIDC);
    long long row = idx / GHIDC;
    float mg = gz[row * (long long)(2 * GHIDC) + j];
    float mx = gz[row * (long long)(2 * GHIDC) + GHIDC + j];
    mact[idx] = (__bf16)(gelu_exact(mg) * mx);
}

// ---------------------------------------------------------------------------
// Launch
// ---------------------------------------------------------------------------
extern "C" void kernel_launch(void* const* d_in, const int* in_sizes, int n_in,
                              void* d_out, int out_size, void* d_ws, size_t ws_size,
                              hipStream_t stream) {
    const float* x           = (const float*)d_in[0];
    const float* W_in        = (const float*)d_in[1];
    const float* conv_w      = (const float*)d_in[2];
    const float* conv_b      = (const float*)d_in[3];
    const float* W_gates     = (const float*)d_in[4];
    const float* b_gates     = (const float*)d_in[5];
    const float* forget_base = (const float*)d_in[6];
    const float* W_out       = (const float*)d_in[7];
    const float* gamma1      = (const float*)d_in[8];
    const float* gamma2      = (const float*)d_in[9];
    const float* W_grow      = (const float*)d_in[10];
    const float* W_shrink    = (const float*)d_in[11];
    float* out = (float*)d_out;

    // ---- workspace layout (bytes) ----
    char* ws = (char*)d_ws;
    size_t off = 0;
    auto alloc = [&](size_t bytes) { char* p = ws + off; off += (bytes + 255) & ~(size_t)255; return p; };
    __bf16* Win_bf  = (__bf16*)alloc((size_t)2*HIDC  * DIMC * 2);
    __bf16* Wg_bf   = (__bf16*)alloc((size_t)2*HIDC  * HIDC * 2);
    __bf16* Wo_bf   = (__bf16*)alloc((size_t)DIMC    * HIDC * 2);
    __bf16* Wgr_bf  = (__bf16*)alloc((size_t)2*GHIDC * DIMC * 2);
    __bf16* Ws_bf   = (__bf16*)alloc((size_t)DIMC    * GHIDC * 2);
    __bf16* xn_bf   = (__bf16*)alloc((size_t)MROWS * DIMC  * 2);
    float*  z       = (float*) alloc((size_t)MROWS * 2*HIDC * 4);   // reused for gz
    float*  xc      = (float*) alloc((size_t)MROWS * HIDC  * 4);
    __bf16* xc_bf   = (__bf16*)alloc((size_t)MROWS * HIDC  * 2);
    float*  g       = (float*) alloc((size_t)MROWS * 2*HIDC * 4);
    __bf16* a_bf    = (__bf16*)alloc((size_t)MROWS * HIDC  * 2);
    float*  x1      = (float*) alloc((size_t)MROWS * DIMC  * 4);
    __bf16* xn2_bf  = (__bf16*)alloc((size_t)MROWS * DIMC  * 2);
    __bf16* mact_bf = (__bf16*)alloc((size_t)MROWS * GHIDC * 2);
    (void)ws_size;

    auto cdiv = [](long long a, long long b) { return (unsigned)((a + b - 1) / b); };

    // 1) weights -> bf16
    f2bf_kernel<<<cdiv((long long)2*HIDC*DIMC,  256), 256, 0, stream>>>(W_in,     Win_bf, (long long)2*HIDC*DIMC);
    f2bf_kernel<<<cdiv((long long)2*HIDC*HIDC,  256), 256, 0, stream>>>(W_gates,  Wg_bf,  (long long)2*HIDC*HIDC);
    f2bf_kernel<<<cdiv((long long)DIMC*HIDC,    256), 256, 0, stream>>>(W_out,    Wo_bf,  (long long)DIMC*HIDC);
    f2bf_kernel<<<cdiv((long long)2*GHIDC*DIMC, 256), 256, 0, stream>>>(W_grow,   Wgr_bf, (long long)2*GHIDC*DIMC);
    f2bf_kernel<<<cdiv((long long)DIMC*GHIDC,   256), 256, 0, stream>>>(W_shrink, Ws_bf,  (long long)DIMC*GHIDC);

    // 2) xn = rmsnorm(x, gamma1)
    rmsnorm_kernel<<<MROWS, 256, 0, stream>>>(x, gamma1, xn_bf, DIMC);

    // 3) z = xn @ W_in^T   (8192 x 4096 x 1024)
    gemm_bf16_wmma_kernel<<<dim3((2*HIDC)/BN, MROWS/BM), 256, 0, stream>>>(
        xn_bf, Win_bf, nullptr, nullptr, z, MROWS, 2*HIDC, DIMC);

    // 4) depthwise causal conv
    conv_kernel<<<cdiv((long long)MROWS*HIDC, 256), 256, 0, stream>>>(z, conv_w, conv_b, xc, xc_bf);

    // 5) g = xc @ W_gates^T + b_gates   (8192 x 4096 x 2048)
    gemm_bf16_wmma_kernel<<<dim3((2*HIDC)/BN, MROWS/BM), 256, 0, stream>>>(
        xc_bf, Wg_bf, b_gates, nullptr, g, MROWS, 2*HIDC, HIDC);

    // 6) fused RG-LRU scan -> a = gelu(gate) * h
    scan_kernel<<<cdiv((long long)BATCH*HIDC, 256), 256, 0, stream>>>(g, xc, z, forget_base, a_bf);

    // 7) x1 = a @ W_out^T + x   (8192 x 1024 x 2048)
    gemm_bf16_wmma_kernel<<<dim3(DIMC/BN, MROWS/BM), 256, 0, stream>>>(
        a_bf, Wo_bf, nullptr, x, x1, MROWS, DIMC, HIDC);

    // 8) xn2 = rmsnorm(x1, gamma2)
    rmsnorm_kernel<<<MROWS, 256, 0, stream>>>(x1, gamma2, xn2_bf, DIMC);

    // 9) gz = xn2 @ W_grow^T   (8192 x 4096 x 1024)  — reuses z buffer
    gemm_bf16_wmma_kernel<<<dim3((2*GHIDC)/BN, MROWS/BM), 256, 0, stream>>>(
        xn2_bf, Wgr_bf, nullptr, nullptr, z, MROWS, 2*GHIDC, DIMC);

    // 10) mact = gelu(mgate) * mx
    gmlp_gate_kernel<<<cdiv((long long)MROWS*GHIDC, 256), 256, 0, stream>>>(z, mact_bf);

    // 11) out = mact @ W_shrink^T + x1   (8192 x 1024 x 2048)
    gemm_bf16_wmma_kernel<<<dim3(DIMC/BN, MROWS/BM), 256, 0, stream>>>(
        mact_bf, Ws_bf, nullptr, x1, out, MROWS, DIMC, GHIDC);
}
